// Experts_5643587026950
// MI455X (gfx1250) — matrix operand order repaired and checked by reading, not compiled, not run on hardware
//
#include <hip/hip_runtime.h>

// ---------------------------------------------------------------------------
// MoE SwiGLU FFN for gfx1250 (MI455X).
//   out[b,e,c,:] = (silu(x@w1) * (x@w3)) @ w2     per expert e
// Fast path: pre-pass repacks f32 weights -> bf16 WMMA B-operand tiles in d_ws,
// main kernel is fused (h in LDS, out in VGPRs) with pure b128-load B operands
// and ds_load_b128 A operands. Fallback (small ws): inline-converting variant.
// ---------------------------------------------------------------------------

typedef __attribute__((ext_vector_type(16))) __bf16 v16bf;
typedef __attribute__((ext_vector_type(8)))  float  v8f;

#define D_MODEL  1024
#define D_FF     2816
#define NEXP     8
#define CCHUNK   2048     // tokens per expert
#define NBATCH   4

#define BM       64       // rows of x per workgroup
#define FC       256      // F-chunk width (11 chunks of 256 = 2816)
#define XS       (D_MODEL + 8)   // LDS stride (halves) for x tile (16B-aligned rows)
#define HS       (FC + 8)        // LDS stride (halves) for h tile (16B-aligned rows)
#define NTHREADS 512             // 16 waves
#define NCHUNK   (D_FF / FC)     // 11

#define W13_K    (NEXP * D_MODEL)          // 8192 rows (expert-major), tiles align
#define W2_K     (NEXP * D_FF)             // 22528 rows
#define W13_KT   (W13_K / 32)              // 256 k-tiles
#define W2_KT    (W2_K / 32)               // 704 k-tiles
#define W13_ELEMS ((size_t)W13_K * D_FF)   // 23,068,672
#define W2_ELEMS  ((size_t)W2_K * D_MODEL) // 23,068,672
#define WS_NEEDED ((W13_ELEMS * 2 + W2_ELEMS) * sizeof(unsigned short))

// pack two f32 -> packed bf16x2 dword (RNE; a low half, b high half)
static __device__ __forceinline__ unsigned int pack_bf2(float a, float b) {
    unsigned int ua = __builtin_bit_cast(unsigned int, a);
    unsigned int ub = __builtin_bit_cast(unsigned int, b);
    ua = (ua + 0x7FFFu + ((ua >> 16) & 1u)) >> 16;
    ub = (ub + 0x7FFFu + ((ub >> 16) & 1u)) & 0xFFFF0000u;
    return ua | ub;
}

static __device__ __forceinline__ unsigned int f2bf(float f) {
    unsigned int x = __builtin_bit_cast(unsigned int, f);
    x = x + 0x7FFFu + ((x >> 16) & 1u);
    return x >> 16;
}

static __device__ __forceinline__ v8f wmma_bf16(v16bf a, v16bf b, v8f c) {
    return __builtin_amdgcn_wmma_f32_16x16x32_bf16(
        false, a, false, b, (short)0, c, false, false);
}

// A-operand (16x32 bf16) from an LDS tile: two ds_load_b128.
// Layout: lanes 0-15 -> M=lane, halves 0..15 = K k0..k0+15
//         lanes 16-31 -> M=lane-16, halves 0..15 = K k0+16..k0+31.
static __device__ __forceinline__ v16bf load_a_lds(const unsigned short* base,
                                                   int stride, int row, int k0, int lhi) {
    const int kb = k0 + (lhi << 4);
    const uint4* p = (const uint4*)(base + row * stride + kb);
    union { uint4 q[2]; v16bf v; } t;
    t.q[0] = p[0];   // K kb..kb+7   (16 B)
    t.q[1] = p[1];   // K kb+8..kb+15 (+16 B)
    return t.v;
}

// B-operand from pre-packed bf16 tiles: tile (kt, ntg) is 1KB contiguous,
// lane L owns 32 bytes at offset L*32. Two global_load_b128, zero VALU.
static __device__ __forceinline__ v16bf load_b_packed(const unsigned short* __restrict__ P,
                                                      int kt_stride, int kt, int ntg,
                                                      int lane) {
    const uint4* p = (const uint4*)(P + ((size_t)ntg * kt_stride + kt) * 512 + lane * 16);
    union { uint4 q[2]; v16bf v; } t;
    t.q[0] = p[0];
    t.q[1] = p[1];
    return t.v;
}

// B-operand built directly from f32 global weights (fallback path).
static __device__ __forceinline__ v16bf load_b_global(const float* __restrict__ W,
                                                      int ld, int k0, int col0,
                                                      int l15, int lhi) {
    const float* p = W + (size_t)(k0 + (lhi << 4)) * ld + (col0 + l15);
    union { unsigned int u[8]; v16bf v; } t;
#pragma unroll
    for (int w = 0; w < 8; ++w)
        t.u[w] = pack_bf2(p[(2 * w) * ld], p[(2 * w + 1) * ld]);
    return t.v;
}

// ---------------------------------------------------------------------------
// Pre-pass: repack f32 W[K][N] -> bf16 B-operand tiles.
// dst tile (kt, nt) at halves offset ((nt*(K/32)+kt)*512); within tile,
// lane owns halves [lane*16, lane*16+16): col = nt*16+(lane&15),
// K = kt*32 + (lane>>4)*16 + h.
// ---------------------------------------------------------------------------
__global__ __launch_bounds__(256)
void repack_w_kernel(const float* __restrict__ src, unsigned short* __restrict__ dst,
                     int K, int N) {
    const int gwave = (int)((blockIdx.x * blockDim.x + threadIdx.x) >> 5);
    const int lane  = (int)(threadIdx.x & 31);
    const int ktn   = K >> 5;
    const int ntile = ktn * (N >> 4);
    if (gwave >= ntile) return;
    const int kt = gwave % ktn;
    const int nt = gwave / ktn;
    const int k0 = kt * 32 + ((lane >> 4) << 4);
    const int n  = nt * 16 + (lane & 15);
    const float* p = src + (size_t)k0 * N + n;
    union { unsigned int u[8]; uint4 q[2]; } t;
#pragma unroll
    for (int w = 0; w < 8; ++w)
        t.u[w] = pack_bf2(p[(size_t)(2 * w) * N], p[(size_t)(2 * w + 1) * N]);
    uint4* d = (uint4*)(dst + ((size_t)nt * ktn + kt) * 512 + lane * 16);
    d[0] = t.q[0];
    d[1] = t.q[1];
}

// ---------------------------------------------------------------------------
// Fast fused kernel: B operands from packed bf16 tiles.
// ---------------------------------------------------------------------------
__global__ __launch_bounds__(NTHREADS)
void moe_swiglu_fast_kernel(const float* __restrict__ x,
                            const unsigned short* __restrict__ pw1,
                            const unsigned short* __restrict__ pw3,
                            const unsigned short* __restrict__ pw2,
                            float* __restrict__ out) {
    extern __shared__ unsigned short smem[];
    unsigned short* xs = smem;              // [BM][XS] bf16 x tile
    unsigned short* hs = smem + BM * XS;    // 2 x [BM][HS] bf16 h chunk

    const int tile = blockIdx.x;
    const int e    = blockIdx.y;
    const int b    = blockIdx.z;
    const int tid  = threadIdx.x;
    const int wave = tid >> 5;
    const int lane = tid & 31;
    const int l15  = lane & 15;
    const int lhi  = lane >> 4;

    const size_t row0  = (size_t)b * (NEXP * CCHUNK) + (size_t)e * CCHUNK + (size_t)tile * BM;
    const float* xbase = x + row0 * D_MODEL;
    const int kt_e1 = e * (D_MODEL / 32);   // k-tile base into packed w1/w3
    const int kt_e2 = e * (D_FF / 32);      // k-tile base into packed w2

    // stage x tile -> LDS bf16
    for (int i = tid; i < BM * (D_MODEL / 4); i += NTHREADS) {
        const int r  = i >> 8;
        const int c4 = i & 255;
        float4 v = ((const float4*)(xbase + (size_t)r * D_MODEL))[c4];
        unsigned int* dst = (unsigned int*)xs + ((r * XS) >> 1) + c4 * 2;
        dst[0] = pack_bf2(v.x, v.y);
        dst[1] = pack_bf2(v.z, v.w);
    }
    __syncthreads();

    const int wm = wave >> 3;   // 0..1 : 32-row half
    const int wn = wave & 7;    // 0..7 : 128-col slice
    v8f oacc[2][8];
#pragma unroll
    for (int mb = 0; mb < 2; ++mb)
#pragma unroll
        for (int nb = 0; nb < 8; ++nb)
            oacc[mb][nb] = (v8f)0.0f;

    for (int ch = 0; ch < NCHUNK; ++ch) {
        const int f0 = ch * FC;
        unsigned short* hbuf = hs + (ch & 1) * (BM * HS);

        // ---- phase 1: gate/up for this wave's 16 F-columns (K = D_MODEL) ----
        v8f g[4], u[4];
#pragma unroll
        for (int rb = 0; rb < 4; ++rb) { g[rb] = (v8f)0.0f; u[rb] = (v8f)0.0f; }

        const int ntg1 = ch * (FC / 16) + wave;     // packed col-group in w1/w3
#pragma unroll 2
        for (int k = 0; k < D_MODEL; k += 32) {
            const int kt = kt_e1 + (k >> 5);
            v16bf bg = load_b_packed(pw1, W13_KT, kt, ntg1, lane);
            v16bf bu = load_b_packed(pw3, W13_KT, kt, ntg1, lane);
#pragma unroll
            for (int rb = 0; rb < 4; ++rb) {
                v16bf a = load_a_lds(xs, XS, rb * 16 + l15, k, lhi);
                g[rb] = wmma_bf16(a, bg, g[rb]);
                u[rb] = wmma_bf16(a, bu, u[rb]);
            }
        }

        // silu(gate) * up -> bf16 h chunk in LDS
        const int hcol = wave * 16 + l15;
#pragma unroll
        for (int rb = 0; rb < 4; ++rb) {
            const int rbase = rb * 16 + (lhi << 3);
#pragma unroll
            for (int i = 0; i < 8; ++i) {
                float gv = g[rb][i];
                float uv = u[rb][i];
                float hv = (gv / (1.0f + __expf(-gv))) * uv;
                hbuf[(rbase + i) * HS + hcol] = (unsigned short)f2bf(hv);
            }
        }
        __syncthreads();   // double buffering covers WAR across chunks

        // ---- phase 2: out += h_chunk @ w2[f0:f0+FC, :] ----
#pragma unroll 2
        for (int k = 0; k < FC; k += 32) {
            const int kt2 = kt_e2 + ((f0 + k) >> 5);
            v16bf a0 = load_a_lds(hbuf, HS, wm * 32 + l15,      k, lhi);
            v16bf a1 = load_a_lds(hbuf, HS, wm * 32 + 16 + l15, k, lhi);
#pragma unroll
            for (int nb = 0; nb < 8; ++nb) {
                v16bf bw = load_b_packed(pw2, W2_KT, kt2, wn * 8 + nb, lane);
                oacc[0][nb] = wmma_bf16(a0, bw, oacc[0][nb]);
                oacc[1][nb] = wmma_bf16(a1, bw, oacc[1][nb]);
            }
        }
    }

#pragma unroll
    for (int mb = 0; mb < 2; ++mb)
#pragma unroll
        for (int nb = 0; nb < 8; ++nb)
#pragma unroll
            for (int i = 0; i < 8; ++i) {
                const int r = wm * 32 + mb * 16 + i + (lhi << 3);
                const int c = wn * 128 + nb * 16 + l15;
                out[(row0 + r) * D_MODEL + c] = oacc[mb][nb][i];
            }
}

// ---------------------------------------------------------------------------
// Fallback fused kernel (no workspace): inline f32->bf16 B operands.
// ---------------------------------------------------------------------------
__global__ __launch_bounds__(NTHREADS)
void moe_swiglu_fallback_kernel(const float* __restrict__ x,
                                const float* __restrict__ w1,
                                const float* __restrict__ w3,
                                const float* __restrict__ w2,
                                float* __restrict__ out) {
    extern __shared__ unsigned short smem[];
    unsigned short* xs = smem;
    unsigned short* hs = smem + BM * XS;

    const int tile = blockIdx.x;
    const int e    = blockIdx.y;
    const int b    = blockIdx.z;
    const int tid  = threadIdx.x;
    const int wave = tid >> 5;
    const int lane = tid & 31;
    const int l15  = lane & 15;
    const int lhi  = lane >> 4;

    const size_t row0  = (size_t)b * (NEXP * CCHUNK) + (size_t)e * CCHUNK + (size_t)tile * BM;
    const float* xbase = x + row0 * D_MODEL;
    const float* w1e   = w1 + (size_t)e * D_MODEL * D_FF;
    const float* w3e   = w3 + (size_t)e * D_MODEL * D_FF;
    const float* w2e   = w2 + (size_t)e * D_FF * D_MODEL;

    for (int i = tid; i < BM * (D_MODEL / 4); i += NTHREADS) {
        const int r  = i >> 8;
        const int c4 = i & 255;
        float4 v = ((const float4*)(xbase + (size_t)r * D_MODEL))[c4];
        unsigned int* dst = (unsigned int*)xs + ((r * XS) >> 1) + c4 * 2;
        dst[0] = pack_bf2(v.x, v.y);
        dst[1] = pack_bf2(v.z, v.w);
    }
    __syncthreads();

    const int wm = wave >> 3;
    const int wn = wave & 7;
    v8f oacc[2][8];
#pragma unroll
    for (int mb = 0; mb < 2; ++mb)
#pragma unroll
        for (int nb = 0; nb < 8; ++nb)
            oacc[mb][nb] = (v8f)0.0f;

    for (int ch = 0; ch < NCHUNK; ++ch) {
        const int f0 = ch * FC;
        unsigned short* hbuf = hs + (ch & 1) * (BM * HS);

        v8f g[4], u[4];
#pragma unroll
        for (int rb = 0; rb < 4; ++rb) { g[rb] = (v8f)0.0f; u[rb] = (v8f)0.0f; }

        const int bcol = f0 + wave * 16;
#pragma unroll 2
        for (int k = 0; k < D_MODEL; k += 32) {
            v16bf bg = load_b_global(w1e, D_FF, k, bcol, l15, lhi);
            v16bf bu = load_b_global(w3e, D_FF, k, bcol, l15, lhi);
#pragma unroll
            for (int rb = 0; rb < 4; ++rb) {
                v16bf a = load_a_lds(xs, XS, rb * 16 + l15, k, lhi);
                g[rb] = wmma_bf16(a, bg, g[rb]);
                u[rb] = wmma_bf16(a, bu, u[rb]);
            }
        }

        const int hcol = wave * 16 + l15;
#pragma unroll
        for (int rb = 0; rb < 4; ++rb) {
            const int rbase = rb * 16 + (lhi << 3);
#pragma unroll
            for (int i = 0; i < 8; ++i) {
                float gv = g[rb][i];
                float uv = u[rb][i];
                float hv = (gv / (1.0f + __expf(-gv))) * uv;
                hbuf[(rbase + i) * HS + hcol] = (unsigned short)f2bf(hv);
            }
        }
        __syncthreads();

#pragma unroll 2
        for (int k = 0; k < FC; k += 32) {
            v16bf a0 = load_a_lds(hbuf, HS, wm * 32 + l15,      k, lhi);
            v16bf a1 = load_a_lds(hbuf, HS, wm * 32 + 16 + l15, k, lhi);
#pragma unroll
            for (int nb = 0; nb < 8; ++nb) {
                v16bf bw = load_b_global(w2e, D_MODEL, f0 + k, wn * 128 + nb * 16, l15, lhi);
                oacc[0][nb] = wmma_bf16(a0, bw, oacc[0][nb]);
                oacc[1][nb] = wmma_bf16(a1, bw, oacc[1][nb]);
            }
        }
    }

#pragma unroll
    for (int mb = 0; mb < 2; ++mb)
#pragma unroll
        for (int nb = 0; nb < 8; ++nb)
#pragma unroll
            for (int i = 0; i < 8; ++i) {
                const int r = wm * 32 + mb * 16 + i + (lhi << 3);
                const int c = wn * 128 + nb * 16 + l15;
                out[(row0 + r) * D_MODEL + c] = oacc[mb][nb][i];
            }
}

extern "C" void kernel_launch(void* const* d_in, const int* in_sizes, int n_in,
                              void* d_out, int out_size, void* d_ws, size_t ws_size,
                              hipStream_t stream) {
    (void)in_sizes; (void)n_in; (void)out_size;
    const float* x  = (const float*)d_in[0];
    const float* w1 = (const float*)d_in[1];
    const float* w3 = (const float*)d_in[2];
    const float* w2 = (const float*)d_in[3];
    float* out = (float*)d_out;

    const size_t smem_bytes = (size_t)(BM * XS + 2 * BM * HS) * sizeof(unsigned short);
    dim3 grid(CCHUNK / BM, NEXP, NBATCH);   // (32, 8, 4)
    dim3 block(NTHREADS);

    if (d_ws != nullptr && ws_size >= WS_NEEDED) {
        unsigned short* pw1 = (unsigned short*)d_ws;
        unsigned short* pw3 = pw1 + W13_ELEMS;
        unsigned short* pw2 = pw3 + W13_ELEMS;

        // repack: 8 waves / block, one 32x16 tile per wave
        const int t13 = W13_KT * (D_FF / 16);    // 45056 tiles
        const int t2  = W2_KT  * (D_MODEL / 16); // 45056 tiles
        repack_w_kernel<<<dim3((t13 + 7) / 8), dim3(256), 0, stream>>>(w1, pw1, W13_K, D_FF);
        repack_w_kernel<<<dim3((t13 + 7) / 8), dim3(256), 0, stream>>>(w3, pw3, W13_K, D_FF);
        repack_w_kernel<<<dim3((t2 + 7) / 8),  dim3(256), 0, stream>>>(w2, pw2, W2_K, D_MODEL);

        moe_swiglu_fast_kernel<<<grid, block, smem_bytes, stream>>>(x, pw1, pw3, pw2, out);
    } else {
        moe_swiglu_fallback_kernel<<<grid, block, smem_bytes, stream>>>(x, w1, w3, w2, out);
    }
}
